// MAE_Contrastive_51891794870621
// MI455X (gfx1250) — matrix operand order repaired and checked by reading, not compiled, not run on hardware
//
#include <hip/hip_runtime.h>
#include <hip/hip_bf16.h>
#include <stdint.h>

// ---------------------------------------------------------------------------
// MAE-GCN autoencoder forward for MI455X (gfx1250), bf16 WMMA everywhere.
//   ~225 GFLOP, ~1.5 GB HBM traffic -> near the 23.3 TB/s memory roof only if
//   matrix math runs on v_wmma_f32_16x16x32_bf16 (fp32 WMMA would be ~8x off).
// ---------------------------------------------------------------------------

typedef __bf16 bf16_t;
typedef __attribute__((ext_vector_type(16))) __bf16 v16bf;
typedef __attribute__((ext_vector_type(8)))  float  v8f;

#define NROWS 10000
#define FDIM  2000
#define H0DIM 512
#define H1DIM 256
#define ZDIM  128

constexpr int BM = 128, BN = 128, BK = 32;
constexpr int LDA_P = BK + 8;   // pad to break LDS bank conflicts (80B rows)
constexpr int LDB_P = BK + 8;

// A_MODE: 0 = A is bf16 row-major; 1 = A is f32 (convert on load, prefetch);
//         2 = A = f32 * f32 mask (fused x*mask), convert on load.
// B_F32 : true = B is f32 weights (convert on load); false = bf16.
// EPI   : 0 = store bf16 (+optional relu); 1 = store f32; 2 = store f32*mask
//         and also pass the mask through to mask_copy (final decoder stage).
template<int A_MODE, bool B_F32, bool RELU, int EPI>
__global__ void __launch_bounds__(256)
wmma_gemm_kernel(const void* __restrict__ Aptr,
                 const void* __restrict__ A2ptr,
                 const void* __restrict__ Bptr,
                 const float* __restrict__ bias,
                 void* __restrict__ Cptr,
                 const float* __restrict__ epi_mask,
                 float* __restrict__ mask_copy,
                 int M, int Ncols, int K,
                 int lda, int ldb, int ldc)
{
  __shared__ alignas(16) bf16_t sA[BM * LDA_P];  // [row][k]   10.0 KB
  __shared__ alignas(16) bf16_t sB[BN * LDB_P];  // [col][k]   10.0 KB (transposed)

  const int tid  = threadIdx.x;
  const int lane = tid & 31;
  const int wave = tid >> 5;
  const int wm   = wave & 3;       // 4 waves along M  -> 32 rows each
  const int wn   = wave >> 2;      // 2 waves along N  -> 64 cols each
  const int m0   = blockIdx.x * BM;
  const int n0   = blockIdx.y * BN;
  const int ln16 = lane & 15;
  const int kh   = lane >> 4;      // lane half selects K sub-range

  const v8f vzero = {0.f,0.f,0.f,0.f,0.f,0.f,0.f,0.f};
  v8f acc[2][4];
  #pragma unroll
  for (int i = 0; i < 2; ++i)
    #pragma unroll
    for (int j = 0; j < 4; ++j) acc[i][j] = vzero;

  for (int kt = 0; kt < K; kt += BK) {
    __syncthreads();

    // -------- stage A tile: BM x BK bf16, row-major in LDS --------
    if (A_MODE == 0) {
      const bf16_t* A = (const bf16_t*)Aptr;
      #pragma unroll
      for (int i = 0; i < 2; ++i) {
        int idx = tid + i * 256;          // 512 chunks of 8 bf16
        int r   = idx >> 2;
        int c8  = (idx & 3) << 3;
        int gr = m0 + r, gc = kt + c8;
        uint4 v = make_uint4(0u, 0u, 0u, 0u);
        if (gr < M && gc < K) v = *(const uint4*)(A + (size_t)gr * lda + gc);
        *(uint4*)&sA[r * LDA_P + c8] = v;  // byte off = 80r + 2*c8, 16B aligned
      }
    } else {
      const float* A  = (const float*)Aptr;
      const float* A2 = (const float*)A2ptr;
      #pragma unroll
      for (int i = 0; i < 4; ++i) {
        int idx = tid + i * 256;          // 1024 float4 chunks
        int r   = idx >> 3;
        int c4  = (idx & 7) << 2;
        int gr = m0 + r, gc = kt + c4;
        float4 v = make_float4(0.f, 0.f, 0.f, 0.f);
        if (gr < M && gc < K) {
          const float* ap = A + (size_t)gr * lda + gc;
          v = *(const float4*)ap;
          if (A_MODE == 2) {
            float4 mv = *(const float4*)(A2 + (size_t)gr * lda + gc);
            v.x *= mv.x; v.y *= mv.y; v.z *= mv.z; v.w *= mv.w;
          } else if (kt + BK < K) {
            __builtin_prefetch(ap + BK, 0, 1);   // global_prefetch_b8 next k-tile
          }
        }
        bf16_t* d = &sA[r * LDA_P + c4];
        d[0] = (bf16_t)v.x; d[1] = (bf16_t)v.y;
        d[2] = (bf16_t)v.z; d[3] = (bf16_t)v.w;
      }
    }

    // -------- stage B tile: BK x BN, stored transposed [n][k] in LDS --------
    if (B_F32) {
      const float* B = (const float*)Bptr;
      #pragma unroll
      for (int i = 0; i < 4; ++i) {
        int idx = tid + i * 256;
        int kk  = idx >> 5;               // 0..31
        int n4  = (idx & 31) << 2;        // 0..124
        int gk = kt + kk, gn = n0 + n4;
        float4 v = make_float4(0.f, 0.f, 0.f, 0.f);
        if (gk < K && gn < Ncols) v = *(const float4*)(B + (size_t)gk * ldb + gn);
        sB[(n4 + 0) * LDB_P + kk] = (bf16_t)v.x;
        sB[(n4 + 1) * LDB_P + kk] = (bf16_t)v.y;
        sB[(n4 + 2) * LDB_P + kk] = (bf16_t)v.z;
        sB[(n4 + 3) * LDB_P + kk] = (bf16_t)v.w;
      }
    } else {
      const bf16_t* B = (const bf16_t*)Bptr;
      #pragma unroll
      for (int i = 0; i < 2; ++i) {
        int idx = tid + i * 256;
        int kk  = idx >> 4;               // 0..31
        int n8  = (idx & 15) << 3;        // 0..120
        int gk = kt + kk, gn = n0 + n8;
        uint4 v = make_uint4(0u, 0u, 0u, 0u);
        if (gk < K && gn < Ncols) v = *(const uint4*)(B + (size_t)gk * ldb + gn);
        alignas(16) bf16_t tmp[8];
        *(uint4*)tmp = v;
        #pragma unroll
        for (int j = 0; j < 8; ++j) sB[(n8 + j) * LDB_P + kk] = tmp[j];
      }
    }
    __syncthreads();

    // -------- WMMA: one 16x16x32 step per (row-tile, col-tile) --------
    // A frag (16x32 bf16): lane L<16 row=L holds K[kh*8..+7] and K[16+kh*8..+7]
    v16bf afrag[2], bfrag[4];
    #pragma unroll
    for (int rt = 0; rt < 2; ++rt) {
      const bf16_t* base = &sA[(wm * 32 + rt * 16 + ln16) * LDA_P + kh * 8];
      __builtin_memcpy(&afrag[rt], base, 16);
      __builtin_memcpy((char*)&afrag[rt] + 16, base + 16, 16);
    }
    // B frag (32x16 bf16): lane L<16 col=L holds K[kh*8..+7] and K[16+kh*8..+7]
    #pragma unroll
    for (int ct = 0; ct < 4; ++ct) {
      const bf16_t* base = &sB[(wn * 64 + ct * 16 + ln16) * LDB_P + kh * 8];
      __builtin_memcpy(&bfrag[ct], base, 16);
      __builtin_memcpy((char*)&bfrag[ct] + 16, base + 16, 16);
    }
    #pragma unroll
    for (int rt = 0; rt < 2; ++rt)
      #pragma unroll
      for (int ct = 0; ct < 4; ++ct)
        acc[rt][ct] = __builtin_amdgcn_wmma_f32_16x16x32_bf16(
            false, afrag[rt], false, bfrag[ct],
            (short)0, acc[rt][ct], false, false);
  }

  // -------- epilogue: C/D layout -> row = tile_m + kh*8 + r, col = tile_n + ln16
  #pragma unroll
  for (int rt = 0; rt < 2; ++rt) {
    #pragma unroll
    for (int ct = 0; ct < 4; ++ct) {
      int col = n0 + wn * 64 + ct * 16 + ln16;
      if (col >= Ncols) continue;
      float bv = bias ? bias[col] : 0.f;
      #pragma unroll
      for (int r = 0; r < 8; ++r) {
        int row = m0 + wm * 32 + rt * 16 + kh * 8 + r;
        if (row >= M) continue;
        float v = acc[rt][ct][r] + bv;
        if (RELU) v = fmaxf(v, 0.f);
        size_t o = (size_t)row * ldc + col;
        if (EPI == 0) {
          ((bf16_t*)Cptr)[o] = (bf16_t)v;
        } else if (EPI == 1) {
          ((float*)Cptr)[o] = v;
        } else {
          float mk = epi_mask[o];
          ((float*)Cptr)[o] = v * mk;    // x_hat = (h@Wd2 + bd2) * mask
          mask_copy[o] = mk;             // fused mask passthrough output
        }
      }
    }
  }
}

// Row-wise L2 normalize of z (N x 128): one wave per row, shfl_xor reduction.
__global__ void __launch_bounds__(256)
znorm_kernel(const float* __restrict__ zraw, float* __restrict__ zout,
             bf16_t* __restrict__ zbf, int nrows)
{
  int wave = threadIdx.x >> 5;
  int lane = threadIdx.x & 31;
  int row  = blockIdx.x * 8 + wave;
  if (row >= nrows) return;
  const float4 v = *(const float4*)(zraw + (size_t)row * ZDIM + lane * 4);
  float s = v.x * v.x + v.y * v.y + v.z * v.z + v.w * v.w;
  #pragma unroll
  for (int off = 16; off > 0; off >>= 1) s += __shfl_xor(s, off, 32);
  float sc = 1.0f / fmaxf(sqrtf(s), 1e-12f);
  float4 o; o.x = v.x * sc; o.y = v.y * sc; o.z = v.z * sc; o.w = v.w * sc;
  *(float4*)(zout + (size_t)row * ZDIM + lane * 4) = o;
  bf16_t* zb = zbf + (size_t)row * ZDIM + lane * 4;
  zb[0] = (bf16_t)o.x; zb[1] = (bf16_t)o.y;
  zb[2] = (bf16_t)o.z; zb[3] = (bf16_t)o.w;
}

static inline dim3 gemm_grid(int M, int Nc) {
  return dim3((unsigned)((M + BM - 1) / BM), (unsigned)((Nc + BN - 1) / BN));
}

extern "C" void kernel_launch(void* const* d_in, const int* in_sizes, int n_in,
                              void* d_out, int out_size, void* d_ws, size_t ws_size,
                              hipStream_t stream)
{
  const float* x    = (const float*)d_in[0];
  const float* adj  = (const float*)d_in[1];
  const float* mask = (const float*)d_in[2];
  const float* We0  = (const float*)d_in[3];
  const float* be0  = (const float*)d_in[4];
  const float* We1  = (const float*)d_in[5];
  const float* be1  = (const float*)d_in[6];
  const float* Wz   = (const float*)d_in[7];
  const float* bz   = (const float*)d_in[8];
  const float* Wd0  = (const float*)d_in[9];
  const float* bd0  = (const float*)d_in[10];
  const float* Wd1  = (const float*)d_in[11];
  const float* bd1  = (const float*)d_in[12];
  const float* Wd2  = (const float*)d_in[13];
  const float* bd2  = (const float*)d_in[14];

  float* out_z    = (float*)d_out;                      // [N, Z]
  float* out_xhat = out_z    + (size_t)NROWS * ZDIM;    // [N, F]
  float* out_mask = out_xhat + (size_t)NROWS * FDIM;    // [N, F]

  // Workspace layout with reuse (peak ~50.3 MB):
  char* ws = (char*)d_ws;
  bf16_t* bufB = (bf16_t*)(ws + 40000000);   // t0 / t1 / tz  (<=10.24 MB)
  bf16_t* h0   = (bf16_t*)(ws + 0);          // also h1, zraw region
  bf16_t* h1   = (bf16_t*)(ws + 0);
  float*  zraw = (float*) (ws + 0);
  bf16_t* zbf  = (bf16_t*)(ws + 8000000);
  bf16_t* d0   = (bf16_t*)(ws + 16000000);
  bf16_t* d1   = (bf16_t*)(ws + 24000000);
  bf16_t* t0 = bufB, *t1 = bufB, *tz = bufB;

  dim3 blk(256);

  // t0 = bf16((x * mask) @ We0)                 [fused mask, no hx buffer]
  wmma_gemm_kernel<2, true, false, 0><<<gemm_grid(NROWS, H0DIM), blk, 0, stream>>>(
      x, mask, We0, nullptr, t0, nullptr, nullptr,
      NROWS, H0DIM, FDIM, FDIM, H0DIM, H0DIM);
  // h0 = relu(adj @ t0 + be0)
  wmma_gemm_kernel<1, false, true, 0><<<gemm_grid(NROWS, H0DIM), blk, 0, stream>>>(
      adj, nullptr, t0, be0, h0, nullptr, nullptr,
      NROWS, H0DIM, NROWS, NROWS, H0DIM, H0DIM);
  // t1 = h0 @ We1
  wmma_gemm_kernel<0, true, false, 0><<<gemm_grid(NROWS, H1DIM), blk, 0, stream>>>(
      h0, nullptr, We1, nullptr, t1, nullptr, nullptr,
      NROWS, H1DIM, H0DIM, H0DIM, H1DIM, H1DIM);
  // h1 = relu(adj @ t1 + be1)
  wmma_gemm_kernel<1, false, true, 0><<<gemm_grid(NROWS, H1DIM), blk, 0, stream>>>(
      adj, nullptr, t1, be1, h1, nullptr, nullptr,
      NROWS, H1DIM, NROWS, NROWS, H1DIM, H1DIM);
  // tz = h1 @ Wz
  wmma_gemm_kernel<0, true, false, 0><<<gemm_grid(NROWS, ZDIM), blk, 0, stream>>>(
      h1, nullptr, Wz, nullptr, tz, nullptr, nullptr,
      NROWS, ZDIM, H1DIM, H1DIM, ZDIM, ZDIM);
  // zraw = adj @ tz + bz   (f32)
  wmma_gemm_kernel<1, false, false, 1><<<gemm_grid(NROWS, ZDIM), blk, 0, stream>>>(
      adj, nullptr, tz, bz, zraw, nullptr, nullptr,
      NROWS, ZDIM, NROWS, NROWS, ZDIM, ZDIM);
  // z = zraw / max(||zraw||, 1e-12)  -> out_z (f32) and zbf (bf16)
  znorm_kernel<<<dim3((NROWS + 7) / 8), blk, 0, stream>>>(zraw, out_z, zbf, NROWS);
  // d0 = relu(z @ Wd0 + bd0)
  wmma_gemm_kernel<0, true, true, 0><<<gemm_grid(NROWS, H1DIM), blk, 0, stream>>>(
      zbf, nullptr, Wd0, bd0, d0, nullptr, nullptr,
      NROWS, H1DIM, ZDIM, ZDIM, H1DIM, H1DIM);
  // d1 = relu(d0 @ Wd1 + bd1)
  wmma_gemm_kernel<0, true, true, 0><<<gemm_grid(NROWS, H0DIM), blk, 0, stream>>>(
      d0, nullptr, Wd1, bd1, d1, nullptr, nullptr,
      NROWS, H0DIM, H1DIM, H1DIM, H0DIM, H0DIM);
  // x_hat = (d1 @ Wd2 + bd2) * mask ; also writes mask passthrough output
  wmma_gemm_kernel<0, true, false, 2><<<gemm_grid(NROWS, FDIM), blk, 0, stream>>>(
      d1, nullptr, Wd2, bd2, out_xhat, mask, out_mask,
      NROWS, FDIM, H0DIM, H0DIM, FDIM, FDIM);

  (void)in_sizes; (void)n_in; (void)out_size; (void)ws_size;
}
